// ForwardDeformer_3934190043843
// MI455X (gfx1250) — compile-verified, matrix operand order checked by reading, not run on hardware
//
#include <hip/hip_runtime.h>

typedef _Float16 f16;
typedef __attribute__((ext_vector_type(16))) _Float16 v16h;
typedef __attribute__((ext_vector_type(8)))  float    v8f;
typedef __attribute__((ext_vector_type(4)))  unsigned u32x4;
typedef __attribute__((ext_vector_type(4)))  float    f32x4;

union V16H { v16h v; f16 e[16]; u32x4 q[2]; };
union V8F  { v8f v; float e[8]; };
union F4   { f32x4 v; float e[4]; };

#define NWID   128   // hidden width
#define NBONES 24
#define NI     9

// ---------------- workspace byte offsets ----------------
#define WS_W0T_L   0        // f16[128*32]  layer0 weights^T (lbs), K padded 3->32
#define WS_W1T_L   8192     // f16[128*128] layer1 weights^T (lbs)
#define WS_W2T_L   40960    // f16[32*128]  layer2 weights^T (lbs), N padded 24->32
#define WS_W0T_D   49152    // f16[128*32]  disp
#define WS_W1T_D   57344    // f16[128*128] disp
#define WS_W2T_D   90112    // f16[16*128]  disp, N padded 3->16
#define WS_C_L     94208    // f32[128]  lat@W0[3:]+b0  (lbs)
#define WS_C_D     94720    // f32[128]  (disp)
#define WS_INV     95232    // f32[9*16] inverses of init-bone tfs

// ---------------- shared memory byte offsets ----------------
#define SH_W0X  0        // f16[128*32]
#define SH_W1   8192     // f16[128*128]
#define SH_W2   40960    // f16[32*128]
#define SH_C    49152    // f32[128]
#define SH_B1   49664    // f32[128]
#define SH_B2   50176    // f32[32]
#define SH_TF   50304    // f32[24*12]  rows 0..2 of each bone tfs
#define SH_INV  51456    // f32[9*16]
#define SH_WAVE 52032
#define WAVE_SZ 13056
#define WV_ACT0 0        // f16[16*32]
#define WV_ACTA 1024     // f16[16*128]
#define WV_ACTB 5120     // f16[16*128]
#define WV_LOG  9216     // f32[16*32]
#define WV_RED  11264    // f32[32]
#define WV_TP   11392    // f32[2*16*12]
#define SMEM_TOTAL (SH_WAVE + 8 * WAVE_SZ)   // 156480 bytes

// softplus(100 v)/100, branchless with raw hw exp2/log2.
// Uses softplus(z) >= z, so fmax(v, r) selects the identity region;
// z clamped at 20 keeps exp2 finite and r ~= 0.2 there (< v), so fmax is exact.
__device__ __forceinline__ float sp100(float v) {
    const float z = fminf(100.f * v, 20.f);
    const float t = __builtin_amdgcn_exp2f(z * 1.4426950408889634f);
    const float r = __builtin_amdgcn_logf(1.f + t) * 0.006931471805599453f; // ln2/100
    return fmaxf(v, r);
}

// One 16-row GEMM: A = acts[16 x 32*nKT] (f16 LDS), B = wT rows = columns of W.
// Two C accumulators per iteration to hide WMMA->VALU hazards and LDS waits.
// mode 0: bias+softplus -> f16 out; mode 1: bias -> f32 out.
__device__ __forceinline__ void mma_layer(const f16* aIn, int sIn, int nKT,
                                          const f16* wT, int sW, int nNT,
                                          const float* bias, int mode,
                                          f16* outH, int sOH,
                                          float* outF, int sOF, int lane) {
    const int row  = lane & 15;
    const int koff = (lane & 16) ? 8 : 0;   // K half-split per ISA A/B layout
    const int Mb   = (lane & 16) ? 8 : 0;   // C layout: M = r or r+8
    V16H A[4];
    for (int kt = 0; kt < nKT; ++kt) {
        const f16* ap = aIn + row * sIn + kt * 32 + koff;
        A[kt].q[0] = *(const u32x4*)(ap);        // K base..base+7
        A[kt].q[1] = *(const u32x4*)(ap + 16);   // K base+16..base+23
    }
    for (int nt = 0; nt < nNT; nt += 2) {
        const bool two = (nt + 1) < nNT;
        V8F c0 = {}, c1 = {};
        for (int kt = 0; kt < nKT; ++kt) {
            const f16* wp0 = wT + (nt * 16 + row) * sW + kt * 32 + koff;
            V16H B0;
            B0.q[0] = *(const u32x4*)(wp0);
            B0.q[1] = *(const u32x4*)(wp0 + 16);
            c0.v = __builtin_amdgcn_wmma_f32_16x16x32_f16(false, A[kt].v, false, B0.v,
                                                          (short)0, c0.v, false, false);
            if (two) {
                const f16* wp1 = wp0 + 16 * sW;
                V16H B1;
                B1.q[0] = *(const u32x4*)(wp1);
                B1.q[1] = *(const u32x4*)(wp1 + 16);
                c1.v = __builtin_amdgcn_wmma_f32_16x16x32_f16(false, A[kt].v, false, B1.v,
                                                              (short)0, c1.v, false, false);
            }
        }
        const int N0 = nt * 16 + row;
        const float b0v = bias[N0];
        const float b1v = two ? bias[N0 + 16] : 0.f;
#pragma unroll
        for (int r = 0; r < 8; ++r) {
            const int M = r + Mb;
            const float v0 = c0.e[r] + b0v;
            if (mode == 0) outH[M * sOH + N0] = (f16)sp100(v0);
            else           outF[M * sOF + N0] = v0;
            if (two) {
                const float v1 = c1.e[r] + b1v;
                if (mode == 0) outH[M * sOH + N0 + 16] = (f16)sp100(v1);
                else           outF[M * sOF + N0 + 16] = v1;
            }
        }
    }
}

// Full lbs MLP eval + softmax blend; returns T[12] = rows 0..2 of sum_j w_j tfs_j.
__device__ __forceinline__ void eval_lbs(float x0, float x1, float x2, int lane,
                                         f16* act0, f16* actA, f16* actB,
                                         float* sLog, float* sRed, float* sTp,
                                         const f16* sW0x, const f16* sW1, const f16* sW2,
                                         const float* sC, const float* sB1, const float* sB2,
                                         const float* sTf, float* T) {
    const int c = lane & 15, half = lane >> 4;
    if (lane < 16) {
        act0[c * 32 + 0] = (f16)x0;
        act0[c * 32 + 1] = (f16)x1;
        act0[c * 32 + 2] = (f16)x2;
    }
    mma_layer(act0, 32, 1, sW0x, 32, 8, sC, 0, actA, 128, (float*)0, 0, lane);
    mma_layer(actA, 128, 4, sW1, 128, 8, sB1, 0, actB, 128, (float*)0, 0, lane);
    mma_layer(actB, 128, 4, sW2, 128, 2, sB2, 1, (f16*)0, 0, sLog, 32, lane);

    // softmax over 24 logits, split 12/12 across lane halves (vectorized LDS reads)
    const f32x4* lp = (const f32x4*)(sLog + c * 32 + half * 12);
    F4 Lq[3];
    Lq[0].v = lp[0]; Lq[1].v = lp[1]; Lq[2].v = lp[2];
    float l[12];
#pragma unroll
    for (int j = 0; j < 12; ++j) l[j] = 20.f * Lq[j >> 2].e[j & 3];
    float mx = l[0];
#pragma unroll
    for (int j = 1; j < 12; ++j) mx = fmaxf(mx, l[j]);
    sRed[half * 16 + c] = mx;
    mx = fmaxf(sRed[c], sRed[16 + c]);
    float e[12], s = 0.f;
#pragma unroll
    for (int j = 0; j < 12; ++j) { e[j] = __expf(l[j] - mx); s += e[j]; }
    sRed[half * 16 + c] = s;
    s = sRed[c] + sRed[16 + c];
    const float is = 1.f / s;

    // partial T = sum_j w_j * tfs3[j]  (rows of 12 floats, 3x b128 per bone)
    F4 a0 = {}, a1 = {}, a2 = {};
#pragma unroll
    for (int j = 0; j < 12; ++j) {
        const f32x4* rp = (const f32x4*)(sTf + (half * 12 + j) * 12);
        const f32x4 r0 = rp[0], r1 = rp[1], r2 = rp[2];
        a0.v += e[j] * r0;
        a1.v += e[j] * r1;
        a2.v += e[j] * r2;
    }
    f32x4* tp = (f32x4*)(sTp + half * 192 + c * 12);
    tp[0] = a0.v * is; tp[1] = a1.v * is; tp[2] = a2.v * is;
    const f32x4* q0 = (const f32x4*)(sTp + c * 12);
    const f32x4* q1 = (const f32x4*)(sTp + 192 + c * 12);
    F4 t0, t1, t2;
    t0.v = q0[0] + q1[0];
    t1.v = q0[1] + q1[1];
    t2.v = q0[2] + q1[2];
#pragma unroll
    for (int t = 0; t < 4; ++t) { T[t] = t0.e[t]; T[4 + t] = t1.e[t]; T[8 + t] = t2.e[t]; }
}

// ----------------- prep kernel: fold latent, transpose+f16 weights, invert init tfs
__global__ void prep_kernel(const float* __restrict__ tfs, const float* __restrict__ lat,
                            const float* __restrict__ lW0, const float* __restrict__ lb0,
                            const float* __restrict__ lW1, const float* __restrict__ lW2,
                            const float* __restrict__ dW0, const float* __restrict__ db0,
                            const float* __restrict__ dW1, const float* __restrict__ dW2,
                            char* __restrict__ ws, int LAT) {
    f16* W0tL = (f16*)(ws + WS_W0T_L);
    f16* W1tL = (f16*)(ws + WS_W1T_L);
    f16* W2tL = (f16*)(ws + WS_W2T_L);
    f16* W0tD = (f16*)(ws + WS_W0T_D);
    f16* W1tD = (f16*)(ws + WS_W1T_D);
    f16* W2tD = (f16*)(ws + WS_W2T_D);
    float* cL = (float*)(ws + WS_C_L);
    float* cD = (float*)(ws + WS_C_D);
    float* iv = (float*)(ws + WS_INV);
    const int t = threadIdx.x;
    if (t < 128) {
        float s = lb0[t];
        for (int k = 0; k < LAT; ++k) s += lat[k] * lW0[(3 + k) * NWID + t];
        cL[t] = s;
    } else {
        const int n = t - 128;
        float s = db0[n];
        for (int k = 0; k < LAT; ++k) s += lat[k] * dW0[(3 + k) * NWID + n];
        cD[n] = s;
    }
    for (int i = t; i < 128 * 32; i += 256) {
        const int n = i >> 5, k = i & 31;
        W0tL[i] = (f16)(k < 3 ? lW0[k * NWID + n] : 0.f);
        W0tD[i] = (f16)(k < 3 ? dW0[k * NWID + n] : 0.f);
    }
    for (int i = t; i < 128 * 128; i += 256) {
        const int n = i >> 7, k = i & 127;
        W1tL[i] = (f16)lW1[k * NWID + n];
        W1tD[i] = (f16)dW1[k * NWID + n];
    }
    for (int i = t; i < 32 * 128; i += 256) {
        const int n = i >> 7, k = i & 127;
        W2tL[i] = (f16)(n < NBONES ? lW2[k * NBONES + n] : 0.f);
    }
    for (int i = t; i < 16 * 128; i += 256) {
        const int n = i >> 7, k = i & 127;
        W2tD[i] = (f16)(n < 3 ? dW2[k * 3 + n] : 0.f);
    }
    if (t < NI) {
        const int IB[NI] = {0, 1, 2, 4, 5, 16, 17, 18, 19};
        const float* A0 = tfs + IB[t] * 16;
        float a[4][4], v[4][4];
        for (int i = 0; i < 4; ++i)
            for (int j = 0; j < 4; ++j) { a[i][j] = A0[i * 4 + j]; v[i][j] = (i == j) ? 1.f : 0.f; }
        for (int col = 0; col < 4; ++col) {
            const float pv = 1.f / a[col][col];
            for (int j = 0; j < 4; ++j) { a[col][j] *= pv; v[col][j] *= pv; }
            for (int r = 0; r < 4; ++r) if (r != col) {
                const float f = a[r][col];
                for (int j = 0; j < 4; ++j) { a[r][j] -= f * a[col][j]; v[r][j] -= f * v[col][j]; }
            }
        }
        for (int i = 0; i < 16; ++i) iv[t * 16 + i] = v[i >> 2][i & 3];
    }
}

// ----------------- main kernel -----------------
__global__ void __launch_bounds__(256) deformer_kernel(
    const float* __restrict__ xd, const float* __restrict__ tfs,
    const float* __restrict__ lbs_b1, const float* __restrict__ lbs_b2,
    const float* __restrict__ disp_b1, const float* __restrict__ disp_b2,
    const char* __restrict__ ws, float* __restrict__ out, int P) {
    extern __shared__ char smem[];
    f16*   sW0x = (f16*)(smem + SH_W0X);
    f16*   sW1  = (f16*)(smem + SH_W1);
    f16*   sW2  = (f16*)(smem + SH_W2);
    float* sC   = (float*)(smem + SH_C);
    float* sB1  = (float*)(smem + SH_B1);
    float* sB2  = (float*)(smem + SH_B2);
    float* sTf  = (float*)(smem + SH_TF);
    float* sInv = (float*)(smem + SH_INV);
    const int tid  = threadIdx.x;
    const int wid  = tid >> 5;
    const int lane = tid & 31;
    char* wb = smem + SH_WAVE + wid * WAVE_SZ;
    f16*   act0 = (f16*)(wb + WV_ACT0);
    f16*   actA = (f16*)(wb + WV_ACTA);
    f16*   actB = (f16*)(wb + WV_ACTB);
    float* sLog = (float*)(wb + WV_LOG);
    float* sRed = (float*)(wb + WV_RED);
    float* sTp  = (float*)(wb + WV_TP);

    // ---- stage lbs weights / constants ----
    const unsigned* wsu = (const unsigned*)ws;
    for (int i = tid; i < 2048; i += 256) ((unsigned*)sW0x)[i] = wsu[WS_W0T_L / 4 + i];
    for (int i = tid; i < 8192; i += 256) ((unsigned*)sW1)[i]  = wsu[WS_W1T_L / 4 + i];
    for (int i = tid; i < 2048; i += 256) ((unsigned*)sW2)[i]  = wsu[WS_W2T_L / 4 + i];
    if (tid < 128) sC[tid]  = ((const float*)(ws + WS_C_L))[tid];
    if (tid < 128) sB1[tid] = lbs_b1[tid];
    if (tid < 32)  sB2[tid] = (tid < NBONES) ? lbs_b2[tid] : 0.f;
    for (int i = tid; i < NBONES * 12; i += 256) { const int j = i / 12, e2 = i % 12; sTf[i] = tfs[j * 16 + e2]; }
    if (tid < NI * 16) sInv[tid] = ((const float*)(ws + WS_INV))[tid];
    __syncthreads();

    // per-wave act0 zero (cols 3..31 stay 0 forever)
    for (int i = lane; i < 16 * 32; i += 32) act0[i] = (f16)0.f;

    // ---- candidate setup ----
    const int PI = P * NI;
    const int c  = lane & 15;
    const int tile = blockIdx.x * 8 + wid;
    const int ci   = tile * 16 + c;
    const bool live = (ci < PI);
    const int cig = live ? ci : (PI - 1);
    const int p  = cig / NI;
    const int ib = cig - p * NI;
    const float xd0 = xd[p * 3 + 0], xd1 = xd[p * 3 + 1], xd2 = xd[p * 3 + 2];
    const float* Mi = sInv + ib * 16;
    float x0 = Mi[0] * xd0 + Mi[1] * xd1 + Mi[2]  * xd2 + Mi[3];
    float x1 = Mi[4] * xd0 + Mi[5] * xd1 + Mi[6]  * xd2 + Mi[7];
    float x2 = Mi[8] * xd0 + Mi[9] * xd1 + Mi[10] * xd2 + Mi[11];

    float T[12];
    eval_lbs(x0, x1, x2, lane, act0, actA, actB, sLog, sRed, sTp,
             sW0x, sW1, sW2, sC, sB1, sB2, sTf, T);
    float g0 = T[0] * x0 + T[1] * x1 + T[2]  * x2 + T[3]  - xd0;
    float g1 = T[4] * x0 + T[5] * x1 + T[6]  * x2 + T[7]  - xd1;
    float g2 = T[8] * x0 + T[9] * x1 + T[10] * x2 + T[11] - xd2;
    // Jinv0 = inverse of T[:3,:3]
    float Ji[9];
    {
        const float a00 = T[0], a01 = T[1], a02 = T[2];
        const float a10 = T[4], a11 = T[5], a12 = T[6];
        const float a20 = T[8], a21 = T[9], a22 = T[10];
        const float c00 = a11 * a22 - a12 * a21, c01 = a02 * a21 - a01 * a22, c02 = a01 * a12 - a02 * a11;
        const float c10 = a12 * a20 - a10 * a22, c11 = a00 * a22 - a02 * a20, c12 = a02 * a10 - a00 * a12;
        const float c20 = a10 * a21 - a11 * a20, c21 = a01 * a20 - a00 * a21, c22 = a00 * a11 - a01 * a10;
        const float id = 1.f / (a00 * c00 + a01 * c10 + a02 * c20);
        Ji[0] = c00 * id; Ji[1] = c01 * id; Ji[2] = c02 * id;
        Ji[3] = c10 * id; Ji[4] = c11 * id; Ji[5] = c12 * id;
        Ji[6] = c20 * id; Ji[7] = c21 * id; Ji[8] = c22 * id;
    }
    float u0 = -(Ji[0] * g0 + Ji[1] * g1 + Ji[2] * g2);
    float u1 = -(Ji[3] * g0 + Ji[4] * g1 + Ji[5] * g2);
    float u2 = -(Ji[6] * g0 + Ji[7] * g1 + Ji[8] * g2);
    float xo0 = x0, xo1 = x1, xo2 = x2;
    float gno = sqrtf(g0 * g0 + g1 * g1 + g2 * g2);
    bool ids = true;

    for (int step = 0; step < 5; ++step) {
        const float dx0 = ids ? u0 : 0.f, dx1 = ids ? u1 : 0.f, dx2 = ids ? u2 : 0.f;
        x0 += dx0; x1 += dx1; x2 += dx2;
        eval_lbs(x0, x1, x2, lane, act0, actA, actB, sLog, sRed, sTp,
                 sW0x, sW1, sW2, sC, sB1, sB2, sTf, T);
        const float gn0 = T[0] * x0 + T[1] * x1 + T[2]  * x2 + T[3]  - xd0;
        const float gn1 = T[4] * x0 + T[5] * x1 + T[6]  * x2 + T[7]  - xd1;
        const float gn2 = T[8] * x0 + T[9] * x1 + T[10] * x2 + T[11] - xd2;
        const float dg0 = ids ? (gn0 - g0) : 0.f;
        const float dg1 = ids ? (gn1 - g1) : 0.f;
        const float dg2 = ids ? (gn2 - g2) : 0.f;
        g0 += dg0; g1 += dg1; g2 += dg2;
        const float gn = sqrtf(g0 * g0 + g1 * g1 + g2 * g2);
        if (gn < gno) { gno = gn; xo0 = x0; xo1 = x1; xo2 = x2; }
        ids = (gno > 1e-5f) && (gn < 1.0f);
        const float vT0 = dx0 * Ji[0] + dx1 * Ji[3] + dx2 * Ji[6];
        const float vT1 = dx0 * Ji[1] + dx1 * Ji[4] + dx2 * Ji[7];
        const float vT2 = dx0 * Ji[2] + dx1 * Ji[5] + dx2 * Ji[8];
        const float a0 = dx0 - (Ji[0] * dg0 + Ji[1] * dg1 + Ji[2] * dg2);
        const float a1 = dx1 - (Ji[3] * dg0 + Ji[4] * dg1 + Ji[5] * dg2);
        const float a2 = dx2 - (Ji[6] * dg0 + Ji[7] * dg1 + Ji[8] * dg2);
        float b = vT0 * dg0 + vT1 * dg1 + vT2 * dg2;
        b += (b >= 0.f) ? 1e-6f : -1e-6f;
        const float ibv = 1.f / b;
        if (ids) {
            Ji[0] += a0 * vT0 * ibv; Ji[1] += a0 * vT1 * ibv; Ji[2] += a0 * vT2 * ibv;
            Ji[3] += a1 * vT0 * ibv; Ji[4] += a1 * vT1 * ibv; Ji[5] += a1 * vT2 * ibv;
            Ji[6] += a2 * vT0 * ibv; Ji[7] += a2 * vT1 * ibv; Ji[8] += a2 * vT2 * ibv;
        }
        u0 = -(Ji[0] * g0 + Ji[1] * g1 + Ji[2] * g2);
        u1 = -(Ji[3] * g0 + Ji[4] * g1 + Ji[5] * g2);
        u2 = -(Ji[6] * g0 + Ji[7] * g1 + Ji[8] * g2);
    }
    const bool valid = (gno < 1e-5f);

    // ---- restage disp weights into the same LDS ----
    __syncthreads();
    for (int i = tid; i < 2048; i += 256) ((unsigned*)sW0x)[i] = wsu[WS_W0T_D / 4 + i];
    for (int i = tid; i < 8192; i += 256) ((unsigned*)sW1)[i]  = wsu[WS_W1T_D / 4 + i];
    for (int i = tid; i < 1024; i += 256) ((unsigned*)sW2)[i]  = wsu[WS_W2T_D / 4 + i];
    if (tid < 128) sC[tid]  = ((const float*)(ws + WS_C_D))[tid];
    if (tid < 128) sB1[tid] = disp_b1[tid];
    if (tid < 32)  sB2[tid] = (tid < 3) ? disp_b2[tid] : 0.f;
    __syncthreads();

    // ---- disp MLP on x_opt ----
    if (lane < 16) {
        act0[c * 32 + 0] = (f16)xo0;
        act0[c * 32 + 1] = (f16)xo1;
        act0[c * 32 + 2] = (f16)xo2;
    }
    mma_layer(act0, 32, 1, sW0x, 32, 8, sC, 0, actA, 128, (float*)0, 0, lane);
    mma_layer(actA, 128, 4, sW1, 128, 8, sB1, 0, actB, 128, (float*)0, 0, lane);
    mma_layer(actB, 128, 4, sW2, 128, 1, sB2, 1, (f16*)0, 0, sLog, 32, lane);

    if (live && lane < 16) {
        out[ci * 3 + 0] = sLog[c * 32 + 0] + xo0;
        out[ci * 3 + 1] = sLog[c * 32 + 1] + xo1;
        out[ci * 3 + 2] = sLog[c * 32 + 2] + xo2;
        out[P * NI * 3 + ci] = valid ? 1.f : 0.f;
    }
}

extern "C" void kernel_launch(void* const* d_in, const int* in_sizes, int n_in,
                              void* d_out, int out_size, void* d_ws, size_t ws_size,
                              hipStream_t stream) {
    const float* xd     = (const float*)d_in[0];
    const float* tfs    = (const float*)d_in[1];
    const float* latent = (const float*)d_in[2];
    const float* lW0 = (const float*)d_in[3];
    const float* lb0 = (const float*)d_in[4];
    const float* lW1 = (const float*)d_in[5];
    const float* lb1 = (const float*)d_in[6];
    const float* lW2 = (const float*)d_in[7];
    const float* lb2 = (const float*)d_in[8];
    const float* dW0 = (const float*)d_in[9];
    const float* db0 = (const float*)d_in[10];
    const float* dW1 = (const float*)d_in[11];
    const float* db1 = (const float*)d_in[12];
    const float* dW2 = (const float*)d_in[13];
    const float* db2 = (const float*)d_in[14];
    const int P   = in_sizes[0] / 3;
    const int LAT = in_sizes[2];

    prep_kernel<<<1, 256, 0, stream>>>(tfs, latent, lW0, lb0, lW1, lW2,
                                       dW0, db0, dW1, dW2, (char*)d_ws, LAT);

    const int tiles  = (P * NI + 15) / 16;
    const int blocks = (tiles + 7) / 8;
    (void)hipFuncSetAttribute(reinterpret_cast<const void*>(deformer_kernel),
                              hipFuncAttributeMaxDynamicSharedMemorySize, SMEM_TOTAL);
    deformer_kernel<<<blocks, 256, SMEM_TOTAL, stream>>>(
        xd, tfs, lb1, lb2, db1, db2, (const char*)d_ws, (float*)d_out, P);
}
